// MOELayer_5385888989892
// MI455X (gfx1250) — matrix operand order repaired and checked by reading, not compiled, not run on hardware
//
#include <hip/hip_runtime.h>
#include <hip/hip_bf16.h>
#include <math.h>

#define S_TOK 16384
#define MDIM  1024
#define HDIM  1024
#define NEXP  64
#define CAP   512   // TOP_K * ceil(S/E) = 2 * 256
#define GTOK  16    // tokens per gate block

typedef __attribute__((ext_vector_type(16))) __bf16 v16bf;
typedef __attribute__((ext_vector_type(8)))  __bf16 v8bf;
typedef __attribute__((ext_vector_type(4)))  __bf16 v4bf;
typedef __attribute__((ext_vector_type(8)))  float  v8f;

// fp32 -> bf16 round-to-nearest-even (bit manipulation)
__device__ __forceinline__ __bf16 f2bf(float f) {
  unsigned u = __builtin_bit_cast(unsigned, f);
  u += 0x7fffu + ((u >> 16) & 1u);
  unsigned short h = (unsigned short)(u >> 16);
  return __builtin_bit_cast(__bf16, h);
}

// two fp32 -> packed bf16 pair (lo = a, hi = b)
__device__ __forceinline__ unsigned packbf(float a, float b) {
  unsigned ua = __builtin_bit_cast(unsigned, a);
  ua += 0x7fffu + ((ua >> 16) & 1u);
  unsigned ub = __builtin_bit_cast(unsigned, b);
  ub += 0x7fffu + ((ub >> 16) & 1u);
  return (ua >> 16) | (ub & 0xffff0000u);
}

// ---------------------------------------------------------------------------
// 1) Gate: 16 tokens per block; wg staged through LDS (16x less L2 traffic
//    than one-block-per-token). Softmax/argmax via wave32 shfl reductions.
// ---------------------------------------------------------------------------
__global__ __launch_bounds__(256) void gate_kernel(
    const float* __restrict__ in, const float* __restrict__ wg,
    float* __restrict__ gates, int* __restrict__ idx1,
    float* __restrict__ gateval) {
  __shared__ float wtile[64 * 68];    // [d-local][e], pad 68 keeps 16B align
  __shared__ float xtile[GTOK * 68];  // [token][d-local]
  __shared__ float lg[GTOK * 64];     // logits
  int s0 = blockIdx.x * GTOK;
  int t = threadIdx.x;
  int e = t & 63;
  int tg = t >> 6;                    // 0..3 -> token subgroup of 4
  float acc0 = 0.f, acc1 = 0.f, acc2 = 0.f, acc3 = 0.f;

  for (int dc = 0; dc < MDIM; dc += 64) {
    #pragma unroll
    for (int i = 0; i < 4; ++i) {     // stage wg[dc..dc+63][:]
      int c = t + i * 256;
      int dl = c >> 4;
      int col = (c & 15) * 4;
      *(float4*)&wtile[dl * 68 + col] =
          *(const float4*)&wg[(size_t)(dc + dl) * NEXP + col];
    }
    {                                 // stage in[s0..s0+15][dc..dc+63]
      int sl = t >> 4;
      int col = (t & 15) * 4;
      *(float4*)&xtile[sl * 68 + col] =
          *(const float4*)&in[(size_t)(s0 + sl) * MDIM + dc + col];
    }
    __syncthreads();
    #pragma unroll 8
    for (int d = 0; d < 64; ++d) {
      float w = wtile[d * 68 + e];
      acc0 = fmaf(xtile[(tg * 4 + 0) * 68 + d], w, acc0);
      acc1 = fmaf(xtile[(tg * 4 + 1) * 68 + d], w, acc1);
      acc2 = fmaf(xtile[(tg * 4 + 2) * 68 + d], w, acc2);
      acc3 = fmaf(xtile[(tg * 4 + 3) * 68 + d], w, acc3);
    }
    __syncthreads();
  }
  lg[(tg * 4 + 0) * 64 + e] = acc0;
  lg[(tg * 4 + 1) * 64 + e] = acc1;
  lg[(tg * 4 + 2) * 64 + e] = acc2;
  lg[(tg * 4 + 3) * 64 + e] = acc3;
  __syncthreads();

  int wv = t >> 5, lane = t & 31;     // 8 waves x 2 tokens each
  #pragma unroll
  for (int ii = 0; ii < 2; ++ii) {
    int s = wv * 2 + ii;
    float l0 = lg[s * 64 + lane];
    float l1 = lg[s * 64 + lane + 32];
    float v; int vi;
    if (l1 > l0) { v = l1; vi = lane + 32; } else { v = l0; vi = lane; }
    #pragma unroll
    for (int off = 16; off > 0; off >>= 1) {   // allreduce max + first-argmax
      float ov = __shfl_xor(v, off, 32);
      int   oi = __shfl_xor(vi, off, 32);
      if (ov > v || (ov == v && oi < vi)) { v = ov; vi = oi; }
    }
    float e0 = expf(l0 - v), e1 = expf(l1 - v);
    float ss = e0 + e1;
    #pragma unroll
    for (int off = 16; off > 0; off >>= 1) ss += __shfl_xor(ss, off, 32);
    float inv = 1.f / ss;                       // gates[argmax] = exp(0)*inv
    gates[(size_t)(s0 + s) * NEXP + lane] = e0 * inv;
    gates[(size_t)(s0 + s) * NEXP + lane + 32] = e1 * inv;
    if (lane == 0) { idx1[s0 + s] = vi; gateval[s0 + s] = inv; }
  }
}

// ---------------------------------------------------------------------------
// 2) me[e] = sum_s gates[s,e]  (deterministic tree reduction per expert)
// ---------------------------------------------------------------------------
__global__ __launch_bounds__(256) void me_kernel(const float* __restrict__ gates,
                                                 float* __restrict__ me) {
  __shared__ float red[256];
  int e = blockIdx.x, t = threadIdx.x;
  float acc = 0.f;
  for (int s = t; s < S_TOK; s += 256) acc += gates[(size_t)s * NEXP + e];
  red[t] = acc; __syncthreads();
  for (int w = 128; w > 0; w >>= 1) { if (t < w) red[t] += red[t + w]; __syncthreads(); }
  if (t == 0) me[e] = red[0];
}

__global__ void init_slots(int* __restrict__ tok) {
  int i = blockIdx.x * blockDim.x + threadIdx.x;
  if (i < NEXP * CAP) tok[i] = -1;
}

// ---------------------------------------------------------------------------
// 3) Ordered per-expert scan (exact cumsum-over-tokens capacity semantics).
// ---------------------------------------------------------------------------
__global__ __launch_bounds__(32) void scan_kernel(const int* __restrict__ idx1,
                                                  int* __restrict__ slot,
                                                  int* __restrict__ tok,
                                                  int* __restrict__ ce) {
  int e = blockIdx.x;
  int lane = threadIdx.x;
  int count = 0;
  unsigned long long lt = (1ull << lane) - 1ull;
  for (int base = 0; base < S_TOK; base += 32) {
    int s = base + lane;
    int mine = (idx1[s] == e) ? 1 : 0;
    unsigned long long m = __ballot(mine);
    if (mine) {
      int loc = count + (int)__popcll(m & lt);
      if (loc < CAP) { slot[s] = loc; tok[e * CAP + loc] = s; }
      else           { slot[s] = -1; }
    }
    count += (int)__popcll(m);
  }
  if (lane == 0) ce[e] = count;
}

__global__ void loss_kernel(const float* __restrict__ me,
                            const int* __restrict__ ce,
                            float* __restrict__ out_loss) {
  if (threadIdx.x == 0) {
    float l = 0.f;
    for (int e = 0; e < NEXP; ++e) l += me[e] * (float)ce[e];
    out_loss[0] = l * ((float)NEXP / ((float)S_TOK * (float)S_TOK));
  }
}

// ---------------------------------------------------------------------------
// 4) Dispatch: gather token rows into [E, CAP, M] bf16 (zero empty slots).
// ---------------------------------------------------------------------------
__global__ __launch_bounds__(256) void dispatch_kernel(const float* __restrict__ in,
                                                       const int* __restrict__ tok,
                                                       __bf16* __restrict__ disp) {
  int slot = blockIdx.x;
  int t = tok[slot];
  int c = threadIdx.x * 4;
  v4bf o;
  if (t >= 0) {
    const float4 v = *(const float4*)&in[(size_t)t * MDIM + c];
    o[0] = f2bf(v.x); o[1] = f2bf(v.y); o[2] = f2bf(v.z); o[3] = f2bf(v.w);
  } else {
    __bf16 z = __builtin_bit_cast(__bf16, (unsigned short)0);
    o[0] = z; o[1] = z; o[2] = z; o[3] = z;
  }
  *(v4bf*)&disp[(size_t)slot * MDIM + c] = o;
}

// ---------------------------------------------------------------------------
// 5) Per-expert tiled GEMM, v_wmma_f32_16x16x32_bf16, double-buffered LDS.
//    A staged with GLOBAL_LOAD_ASYNC_TO_LDS_B128 (ASYNCcnt-tracked, no VGPR
//    transit); B staged f32->bf16 with packed b32 DS stores.
// ---------------------------------------------------------------------------
template <bool RELU_BF16_OUT>
__global__ __launch_bounds__(256) void moe_expert_gemm(
    const __bf16* __restrict__ A,   // [E, CAP, 1024] bf16
    const float*  __restrict__ W,   // [E, 1024, 1024] f32
    __bf16* __restrict__ Dbf,       // used when RELU_BF16_OUT
    float*  __restrict__ Df) {      // used otherwise
  constexpr int BM = 128, BN = 128, BK = 32;
  constexpr int LDA = BK + 8;       // stride 80B -> conflict-free frag reads
  constexpr int LDB = BK + 8;       // B column-major in LDS: [BN][BK]
  __shared__ __bf16 Alds[2][BM * LDA];
  __shared__ __bf16 Blds[2][BN * LDB];

  int z  = blockIdx.z;
  int n0 = blockIdx.x * BN;
  int m0 = blockIdx.y * BM;
  int tid  = threadIdx.x;
  int lane = tid & 31;
  int wave = tid >> 5;
  int wrow = (wave >> 1) * 32;
  int wcol = (wave & 1) * 64;

  const __bf16* Ab = A + ((size_t)z * CAP + m0) * (size_t)MDIM;
  const float*  Wb = W + (size_t)z * MDIM * HDIM;

  v8f acc[2][4] = {};

  auto stage = [&](int k0, int p) {
    // A tile 128x32 bf16: async byte copy straight into LDS
    #pragma unroll
    for (int i = 0; i < 2; ++i) {
      int c = tid + i * 256;               // 0..511
      int r = c >> 2;
      int off = (c & 3) * 8;
      unsigned lds = (unsigned)(size_t)&Alds[p][r * LDA + off];
      unsigned long long ga =
          (unsigned long long)(size_t)&Ab[(size_t)r * MDIM + k0 + off];
      asm volatile("global_load_async_to_lds_b128 %0, %1, off"
                   :: "v"(lds), "v"(ga) : "memory");
    }
    // B tile 32x128 f32 -> bf16, two K-rows per thread, packed b32 stores
    #pragma unroll
    for (int i = 0; i < 2; ++i) {
      int c = tid + i * 256;               // 0..511
      int k = (c >> 5) * 2;                // even k
      int nn = (c & 31) * 4;
      const float4 b0 = *(const float4*)&Wb[(size_t)(k0 + k) * HDIM + n0 + nn];
      const float4 b1 = *(const float4*)&Wb[(size_t)(k0 + k + 1) * HDIM + n0 + nn];
      *(unsigned*)&Blds[p][(nn + 0) * LDB + k] = packbf(b0.x, b1.x);
      *(unsigned*)&Blds[p][(nn + 1) * LDB + k] = packbf(b0.y, b1.y);
      *(unsigned*)&Blds[p][(nn + 2) * LDB + k] = packbf(b0.z, b1.z);
      *(unsigned*)&Blds[p][(nn + 3) * LDB + k] = packbf(b0.w, b1.w);
    }
  };

  auto compute = [&](int p) {
    int arow = lane & 15;
    int ah = (lane >> 4) * 8;   // A: low lanes K0..7/16..23, high K8..15/24..31
    int bh = (lane >> 4) * 8;   // B: low lanes K0..7/16..23, high K8..15/24..31
    #pragma unroll
    for (int i = 0; i < 2; ++i) {
      const __bf16* ap = &Alds[p][(wrow + i * 16 + arow) * LDA];
      v8bf alo = *(const v8bf*)&ap[ah];
      v8bf ahi = *(const v8bf*)&ap[16 + ah];
      v16bf a = __builtin_shufflevector(alo, ahi, 0, 1, 2, 3, 4, 5, 6, 7,
                                        8, 9, 10, 11, 12, 13, 14, 15);
      #pragma unroll
      for (int j = 0; j < 4; ++j) {
        const __bf16* bp = &Blds[p][(wcol + j * 16 + (lane & 15)) * LDB + bh];
        v8bf blo = *(const v8bf*)&bp[0];
        v8bf bhi = *(const v8bf*)&bp[16];
        v16bf b = __builtin_shufflevector(blo, bhi, 0, 1, 2, 3, 4, 5, 6, 7,
                                          8, 9, 10, 11, 12, 13, 14, 15);
        acc[i][j] = __builtin_amdgcn_wmma_f32_16x16x32_bf16(
            false, a, false, b, (short)0, acc[i][j], false, false);
      }
    }
  };

  stage(0, 0);
  asm volatile("s_wait_asynccnt 0x0" ::: "memory");
  __syncthreads();
  int p = 0;
  for (int k0 = BK; k0 < MDIM; k0 += BK) {
    stage(k0, p ^ 1);          // next tile -> other buffer, overlaps WMMAs
    compute(p);
    asm volatile("s_wait_asynccnt 0x0" ::: "memory");
    __syncthreads();
    p ^= 1;
  }
  compute(p);

  // writeback: VGPR r, lanes<16 -> row r, lanes>=16 -> row r+8
  int radd = (lane >> 4) * 8;
  int cadd = lane & 15;
  #pragma unroll
  for (int i = 0; i < 2; ++i) {
    #pragma unroll
    for (int j = 0; j < 4; ++j) {
      #pragma unroll
      for (int r = 0; r < 8; ++r) {
        int row = m0 + wrow + i * 16 + r + radd;
        int col = n0 + wcol + j * 16 + cadd;
        float v = acc[i][j][r];
        size_t o = ((size_t)z * CAP + row) * (size_t)HDIM + col;
        if (RELU_BF16_OUT) {
          Dbf[o] = f2bf(fmaxf(v, 0.f));
        } else {
          Df[o] = v;
        }
      }
    }
  }
}

// ---------------------------------------------------------------------------
// 6) Combine: out[s,:] = gate1[s] * expert_out[idx[s], slot[s], :]
// ---------------------------------------------------------------------------
__global__ __launch_bounds__(256) void combine_kernel(
    const float* __restrict__ eout, const int* __restrict__ idx1,
    const int* __restrict__ slot, const float* __restrict__ gateval,
    float* __restrict__ out) {
  int s = blockIdx.x;
  int c = threadIdx.x * 4;
  int sl = slot[s];
  float4 o;
  if (sl >= 0) {
    int e = idx1[s];
    float g = gateval[s];
    const float4 v = *(const float4*)&eout[((size_t)e * CAP + sl) * MDIM + c];
    o.x = g * v.x; o.y = g * v.y; o.z = g * v.z; o.w = g * v.w;
  } else {
    o.x = 0.f; o.y = 0.f; o.z = 0.f; o.w = 0.f;
  }
  *(float4*)&out[(size_t)s * MDIM + c] = o;
}

// ---------------------------------------------------------------------------
extern "C" void kernel_launch(void* const* d_in, const int* in_sizes, int n_in,
                              void* d_out, int out_size, void* d_ws, size_t ws_size,
                              hipStream_t stream) {
  (void)in_sizes; (void)n_in; (void)out_size; (void)ws_size;
  const float* in = (const float*)d_in[0];
  const float* wg = (const float*)d_in[1];
  const float* w1 = (const float*)d_in[2];
  const float* w2 = (const float*)d_in[3];
  float* out = (float*)d_out;

  char* ws = (char*)d_ws;
  size_t off = 0;
  auto alloc = [&](size_t bytes) -> void* {
    void* p = ws + off;
    off = (off + bytes + 255) & ~(size_t)255;
    return p;
  };
  float*  gates   = (float*)alloc((size_t)S_TOK * NEXP * 4);          // 4 MB
  float*  me      = (float*)alloc(NEXP * 4);
  int*    ce      = (int*)  alloc(NEXP * 4);
  int*    idx1    = (int*)  alloc(S_TOK * 4);
  float*  gateval = (float*)alloc(S_TOK * 4);
  int*    slot    = (int*)  alloc(S_TOK * 4);
  int*    tok     = (int*)  alloc((size_t)NEXP * CAP * 4);
  __bf16* disp    = (__bf16*)alloc((size_t)NEXP * CAP * MDIM * 2);    // 64 MB
  __bf16* hbuf    = (__bf16*)alloc((size_t)NEXP * CAP * HDIM * 2);    // 64 MB
  float*  eout    = (float*)alloc((size_t)NEXP * CAP * MDIM * 4);     // 128 MB

  gate_kernel<<<S_TOK / GTOK, 256, 0, stream>>>(in, wg, gates, idx1, gateval);
  me_kernel<<<NEXP, 256, 0, stream>>>(gates, me);
  init_slots<<<(NEXP * CAP + 255) / 256, 256, 0, stream>>>(tok);
  scan_kernel<<<NEXP, 32, 0, stream>>>(idx1, slot, tok, ce);
  loss_kernel<<<1, 32, 0, stream>>>(me, ce, out + (size_t)S_TOK * MDIM);
  dispatch_kernel<<<NEXP * CAP, 256, 0, stream>>>(in, tok, disp);

  dim3 g(HDIM / 128, CAP / 128, NEXP);
  moe_expert_gemm<true ><<<g, 256, 0, stream>>>(disp, w1, hbuf, nullptr);
  moe_expert_gemm<false><<<g, 256, 0, stream>>>(hbuf, w2, nullptr, eout);

  combine_kernel<<<S_TOK, 256, 0, stream>>>(eout, idx1, slot, gateval, out);
}
